// ToposMambaBlock_61512521613804
// MI455X (gfx1250) — compile-verified, matrix operand order checked by reading, not run on hardware
//
#include <hip/hip_runtime.h>
#include <hip/hip_bf16.h>
#include <math.h>

// ---------------------------------------------------------------------------
// ToposMambaBlock for MI455X (gfx1250, wave32, WMMA)
//   B=4, S=512, D=768, N=64
//   out = [ clip(y+x) : B*S*D floats | h_final : B*D*N floats ]
// ---------------------------------------------------------------------------

typedef __attribute__((ext_vector_type(16))) _Float16 v16h;
typedef __attribute__((ext_vector_type(8)))  _Float16 v8h;
typedef __attribute__((ext_vector_type(8)))  float    v8f;
typedef __attribute__((ext_vector_type(4)))  float    v4f;

#define BSZ  4
#define SSEQ 512
#define DD   768
#define NN   64
#define LN_EPS 1e-5f

__device__ __forceinline__ float sigmoidf_(float v) {
    return 1.0f / (1.0f + __expf(-v));
}

// ---------------------------------------------------------------------------
// Kernel 0: sigmoid all weights once; convert x and sigmoid(Braw) to f16
// ---------------------------------------------------------------------------
__global__ void topos_prep_kernel(const float* __restrict__ x,
                                  const float* __restrict__ Araw,
                                  const float* __restrict__ Braw,
                                  const float* __restrict__ Craw,
                                  const float* __restrict__ graw,
                                  _Float16* __restrict__ Xh,
                                  _Float16* __restrict__ Bwh,
                                  float* __restrict__ Agate,
                                  float* __restrict__ Cw,
                                  float* __restrict__ gamma) {
    int i = blockIdx.x * blockDim.x + threadIdx.x;
    if (i < BSZ * SSEQ * DD) Xh[i] = (_Float16)x[i];
    if (i < DD * NN) {
        Agate[i] = sigmoidf_(Araw[i]);
        Cw[i]    = sigmoidf_(Craw[i]);
        Bwh[i]   = (_Float16)sigmoidf_(Braw[i]);   // N*D == D*N elements
    }
    if (i < DD) gamma[i] = sigmoidf_(graw[i]);
}

// ---------------------------------------------------------------------------
// Kernel 1: U[m, n] = sum_k Xh[m, k] * Bwh[n, k]   (m = b*S+t, k = d)
//   M=2048, K=768, N=64.  One wave per 16x16 tile, v_wmma_f32_16x16x32_f16.
//   Fragment layouts per CDNA5 ISA 7.12.2:
//     A (16x32 f16): lane<16 -> M=lane, halves 0..7 = K 0..7, 8..15 = K 16..23
//                    lane>=16 -> same M, K bases +8
//     D (16x16 f32): lane gives N = lane&15, elem r -> M = r + (lane>>4)*8
// ---------------------------------------------------------------------------
__global__ void topos_u_gemm_kernel(const _Float16* __restrict__ Xh,
                                    const _Float16* __restrict__ Bwh,
                                    float* __restrict__ U) {
    const int M  = BSZ * SSEQ;          // 2048
    const int NT = NN / 16;             // 4 tiles in N
    int wave = (blockIdx.x * blockDim.x + threadIdx.x) >> 5;
    int lane = threadIdx.x & 31;
    int mt = wave / NT;
    int nt = wave % NT;
    if (mt >= M / 16) return;

    int row  = lane & 15;               // M index for A, N index for B
    int ksel = lane >> 4;

    const _Float16* arow = Xh  + (size_t)(mt * 16 + row) * DD;
    const _Float16* brow = Bwh + (size_t)(nt * 16 + row) * DD;

    v8f acc = {};
    for (int k = 0; k < DD; k += 32) {
        if (k + 64 < DD) {              // speculative prefetch of next tiles
            __builtin_prefetch(arow + k + 64, 0, 1);
            __builtin_prefetch(brow + k + 64, 0, 1);
        }
        v8h alo = *(const v8h*)(arow + k + ksel * 8);
        v8h ahi = *(const v8h*)(arow + k + 16 + ksel * 8);
        v8h blo = *(const v8h*)(brow + k + ksel * 8);
        v8h bhi = *(const v8h*)(brow + k + 16 + ksel * 8);
        v16h a, b;
#pragma unroll
        for (int e = 0; e < 8; ++e) {
            a[e] = alo[e]; a[e + 8] = ahi[e];
            b[e] = blo[e]; b[e + 8] = bhi[e];
        }
        // (neg_a, A, neg_b, B, c_mod, C, reuse_a, reuse_b)
        acc = __builtin_amdgcn_wmma_f32_16x16x32_f16(
            false, a, false, b, (short)0, acc, false, false);
    }

    int n = nt * 16 + (lane & 15);
#pragma unroll
    for (int r = 0; r < 8; ++r) {
        int m = mt * 16 + r + ksel * 8;
        U[(size_t)m * NN + n] = acc[r];
    }
}

// ---------------------------------------------------------------------------
// Kernel 2: the serial scan. One thread per (b, d); h[64], A_gate[64], Cw[64]
// live in VGPRs for all 512 steps. U_t broadcast through LDS. Writes raw
// (pre-layernorm) y into d_out; layernorm deferred to kernel 3 (removes all
// cross-workgroup coupling from the recurrence).
// ---------------------------------------------------------------------------
__global__ void __launch_bounds__(256)
topos_scan_kernel(const float* __restrict__ x,
                  const float* __restrict__ U,
                  const float* __restrict__ Agate,
                  const float* __restrict__ Cw,
                  float* __restrict__ yraw,     // d_out, [B,S,D]
                  float* __restrict__ hfin) {   // d_out + B*S*D, [B,D,N]
    const int DCH = DD / 256;                   // 3 d-chunks per batch
    int b = blockIdx.x / DCH;
    int d = (blockIdx.x % DCH) * 256 + threadIdx.x;

    float h[NN], ag[NN], cw[NN];
    const v4f* agp = (const v4f*)(Agate + (size_t)d * NN);
    const v4f* cwp = (const v4f*)(Cw    + (size_t)d * NN);
#pragma unroll
    for (int q = 0; q < NN / 4; ++q) {
        v4f a4 = agp[q], c4 = cwp[q];
#pragma unroll
        for (int e = 0; e < 4; ++e) {
            ag[q * 4 + e] = a4[e];
            cw[q * 4 + e] = c4[e];
            h [q * 4 + e] = 0.0f;
        }
    }

    __shared__ float uS[NN];
    const float* xb = x    + (size_t)b * SSEQ * DD;
    const float* Ub = U    + (size_t)b * SSEQ * NN;
    float*       yb = yraw + (size_t)b * SSEQ * DD;

    for (int t = 0; t < SSEQ; ++t) {
        if (threadIdx.x < NN) uS[threadIdx.x] = Ub[(size_t)t * NN + threadIdx.x];
        __syncthreads();
        float xv = xb[(size_t)t * DD + d];
        float y = 0.0f;
#pragma unroll
        for (int n = 0; n < NN; ++n) {
            float hv = fmaf(h[n], ag[n], uS[n] * xv);
            hv = fminf(fmaxf(hv, 0.0f), 1.0f);
            h[n] = hv;
            y = fmaf(hv, cw[n], y);
        }
        yb[(size_t)t * DD + d] = y;
        __syncthreads();
    }

#pragma unroll
    for (int n = 0; n < NN; ++n)
        hfin[((size_t)b * DD + d) * NN + n] = h[n];
}

// ---------------------------------------------------------------------------
// Kernel 3: layernorm over D + sigmoid-gamma + residual + clip, in place on
// d_out. One block per (b,t) row: 2048 fully independent blocks.
// ---------------------------------------------------------------------------
__global__ void __launch_bounds__(256)
topos_norm_kernel(const float* __restrict__ x,
                  const float* __restrict__ gamma,
                  float* __restrict__ y) {
    int row = blockIdx.x;                       // b*S + t
    const float* xr = x + (size_t)row * DD;
    float*       yr = y + (size_t)row * DD;
    int tid = threadIdx.x;

    float v0 = yr[tid], v1 = yr[tid + 256], v2 = yr[tid + 512];
    __shared__ float s1[256], s2[256];
    s1[tid] = v0 + v1 + v2;
    s2[tid] = v0 * v0 + v1 * v1 + v2 * v2;
    __syncthreads();
    for (int off = 128; off > 0; off >>= 1) {
        if (tid < off) {
            s1[tid] += s1[tid + off];
            s2[tid] += s2[tid + off];
        }
        __syncthreads();
    }
    float mu  = s1[0] * (1.0f / DD);
    float var = s2[0] * (1.0f / DD) - mu * mu;  // population variance (jnp.var)
    float rs  = __frsqrt_rn(var + LN_EPS);

    float o0 = (v0 - mu) * rs * gamma[tid]       + xr[tid];
    float o1 = (v1 - mu) * rs * gamma[tid + 256] + xr[tid + 256];
    float o2 = (v2 - mu) * rs * gamma[tid + 512] + xr[tid + 512];
    yr[tid]       = fminf(fmaxf(o0, 0.0f), 1.0f);
    yr[tid + 256] = fminf(fmaxf(o1, 0.0f), 1.0f);
    yr[tid + 512] = fminf(fmaxf(o2, 0.0f), 1.0f);
}

// ---------------------------------------------------------------------------
// Host-side launcher
// ---------------------------------------------------------------------------
extern "C" void kernel_launch(void* const* d_in, const int* in_sizes, int n_in,
                              void* d_out, int out_size, void* d_ws, size_t ws_size,
                              hipStream_t stream) {
    const float* x    = (const float*)d_in[0];   // [B,S,D]
    const float* Araw = (const float*)d_in[1];   // [D,N]
    const float* Braw = (const float*)d_in[2];   // [N,D]
    const float* Craw = (const float*)d_in[3];   // [D,N]
    const float* graw = (const float*)d_in[4];   // [D]

    float* out  = (float*)d_out;
    float* yout = out;                            // [B,S,D]
    float* hfin = out + (size_t)BSZ * SSEQ * DD;  // [B,D,N]

    // workspace layout (256B aligned slices)
    char* ws = (char*)d_ws;
    size_t off = 0;
    auto take = [&](size_t bytes) {
        char* p = ws + off;
        off += (bytes + 255) & ~(size_t)255;
        return p;
    };
    _Float16* Xh    = (_Float16*)take((size_t)BSZ * SSEQ * DD * sizeof(_Float16));
    _Float16* Bwh   = (_Float16*)take((size_t)NN * DD * sizeof(_Float16));
    float*    Agate = (float*)   take((size_t)DD * NN * sizeof(float));
    float*    Cw    = (float*)   take((size_t)DD * NN * sizeof(float));
    float*    gamma = (float*)   take((size_t)DD * sizeof(float));
    float*    U     = (float*)   take((size_t)BSZ * SSEQ * NN * sizeof(float));

    // 0: weight prep / precision conversion
    {
        int n = BSZ * SSEQ * DD;
        topos_prep_kernel<<<(n + 255) / 256, 256, 0, stream>>>(
            x, Araw, Braw, Craw, graw, Xh, Bwh, Agate, Cw, gamma);
    }
    // 1: U = X @ Bw^T via WMMA (512 waves: 128 blocks x 4 waves)
    {
        int tiles = (BSZ * SSEQ / 16) * (NN / 16);   // 512
        topos_u_gemm_kernel<<<tiles / 4, 128, 0, stream>>>(Xh, Bwh, U);
    }
    // 2: the sequential scan (state in registers, raw y to d_out)
    {
        topos_scan_kernel<<<BSZ * (DD / 256), 256, 0, stream>>>(
            x, U, Agate, Cw, yout, hfin);
    }
    // 3: layernorm + residual + clip, in place
    {
        topos_norm_kernel<<<BSZ * SSEQ, 256, 0, stream>>>(x, gamma, yout);
    }
}
